// GCNClassifier_63522566307870
// MI455X (gfx1250) — compile-verified, matrix operand order checked by reading, not run on hardware
//
#include <hip/hip_runtime.h>
#include <hip/hip_bf16.h>
#include <stdint.h>

typedef __attribute__((ext_vector_type(16))) __bf16 v16bf;
typedef __attribute__((ext_vector_type(8)))  __bf16 v8bf;
typedef __attribute__((ext_vector_type(8)))  float  v8f;

#define BLK 256

// ---------- small elementwise kernels ----------
__global__ void k_fill(float* __restrict__ p, float v, long n) {
  long t = (long)blockIdx.x * blockDim.x + threadIdx.x;
  if (t < n) p[t] = v;
}

__global__ void k_deg_accum(float* __restrict__ deg, const int* __restrict__ dst,
                            const float* __restrict__ ew, int E) {
  int t = blockIdx.x * blockDim.x + threadIdx.x;
  if (t < E) atomicAdd(deg + dst[t], ew[t]);
}

__global__ void k_dinv(float* __restrict__ d, int N) {
  int t = blockIdx.x * blockDim.x + threadIdx.x;
  if (t < N) { float v = d[t]; d[t] = (v > 0.f) ? rsqrtf(v) : 0.f; }
}

// Vectorized f32 -> bf16 conversion (8 elements / thread), optional ReLU.
__global__ void k_cvt8_bf16(const float* __restrict__ in, __bf16* __restrict__ out,
                            long n8, int relu) {
  long t = (long)blockIdx.x * blockDim.x + threadIdx.x;
  if (t >= n8) return;
  const float4* p = (const float4*)in + t * 2;
  float4 a = p[0], b = p[1];
  if (relu) {
    a.x = a.x > 0.f ? a.x : 0.f;  a.y = a.y > 0.f ? a.y : 0.f;
    a.z = a.z > 0.f ? a.z : 0.f;  a.w = a.w > 0.f ? a.w : 0.f;
    b.x = b.x > 0.f ? b.x : 0.f;  b.y = b.y > 0.f ? b.y : 0.f;
    b.z = b.z > 0.f ? b.z : 0.f;  b.w = b.w > 0.f ? b.w : 0.f;
  }
  v8bf o;
  o[0] = (__bf16)a.x; o[1] = (__bf16)a.y; o[2] = (__bf16)a.z; o[3] = (__bf16)a.w;
  o[4] = (__bf16)b.x; o[5] = (__bf16)b.y; o[6] = (__bf16)b.z; o[7] = (__bf16)b.w;
  ((v8bf*)out)[t] = o;
}

__global__ void k_cvt_tail_bf16(const float* __restrict__ in, __bf16* __restrict__ out,
                                long start, long n, int relu) {
  long t = start + (long)blockIdx.x * blockDim.x + threadIdx.x;
  if (t < n) {
    float v = in[t];
    if (relu && v < 0.f) v = 0.f;
    out[t] = (__bf16)v;
  }
}

// Shuffle row-major W[K][Nout] (f32) into per-lane B-matrix WMMA fragments (bf16).
// Fragment index: ((c*T + coltile)*32 + lane)*16 + j, where for lane l:
//   n = l&15, khalf = l>>4, element j covers k = c*32 + (j<8?0:16) + khalf*8 + (j&7)
__global__ void k_wfrag(const float* __restrict__ W, __bf16* __restrict__ Wf, int K, int Nout) {
  int t = blockIdx.x * blockDim.x + threadIdx.x;
  if (t >= K * Nout) return;
  int j  = t & 15;
  int l  = (t >> 4) & 31;
  int g  = t >> 9;               // c*T + coltile
  int T  = Nout >> 4;
  int tc = g % T;
  int c  = g / T;
  int n = l & 15, khalf = l >> 4;
  int k = c * 32 + ((j < 8) ? 0 : 16) + khalf * 8 + (j & 7);
  Wf[t] = (__bf16)W[(size_t)k * Nout + tc * 16 + n];
}

// ---------- WMMA GEMM with fused GCN epilogue ----------
// H[N x Nout] = A[N x K] @ B; also O = bias[col] + H * dinv[row]^2  (self-loop init).
// Block = 32*(Nout/16) threads; each wave computes one column tile for TWO 16-row
// M-tiles, sharing every B-fragment load between them (2 WMMAs per B load).
__global__ void k_wmma_gemm_bf16(const __bf16* __restrict__ A, const __bf16* __restrict__ Bfrag,
                                 const float* __restrict__ bias, const float* __restrict__ dinv,
                                 float* __restrict__ H, float* __restrict__ O,
                                 int Nrows, int K, int Nout) {
  const int lane = threadIdx.x & 31;
  const int wave = threadIdx.x >> 5;          // column tile index
  const int T = Nout >> 4;
  const int rowbase0 = blockIdx.x * 32;
  const int rowbase1 = rowbase0 + 16;
  const int nchunks = K >> 5;

  int ar0 = rowbase0 + (lane & 15); if (ar0 >= Nrows) ar0 = Nrows - 1;  // clamp, no divergence
  int ar1 = rowbase1 + (lane & 15); if (ar1 >= Nrows) ar1 = Nrows - 1;
  const __bf16* ap0 = A + (size_t)ar0 * K + ((lane >> 4) * 8);
  const __bf16* ap1 = A + (size_t)ar1 * K + ((lane >> 4) * 8);

  v8f acc0 = {0.f, 0.f, 0.f, 0.f, 0.f, 0.f, 0.f, 0.f};
  v8f acc1 = {0.f, 0.f, 0.f, 0.f, 0.f, 0.f, 0.f, 0.f};
  for (int c = 0; c < nchunks; ++c) {
    v16bf b = *(const v16bf*)(Bfrag + ((size_t)(c * T + wave) * 32 + lane) * 16);
    v8bf g0 = *(const v8bf*)(ap0 + c * 32);        // k = c*32 + khalf*8 .. +7
    v8bf g1 = *(const v8bf*)(ap0 + c * 32 + 16);   // k = c*32 + 16 + khalf*8 .. +7
    v16bf a0 = __builtin_shufflevector(g0, g1, 0,1,2,3,4,5,6,7,8,9,10,11,12,13,14,15);
    acc0 = __builtin_amdgcn_wmma_f32_16x16x32_bf16(false, a0, false, b, (short)0, acc0, false, false);
    v8bf h0 = *(const v8bf*)(ap1 + c * 32);
    v8bf h1 = *(const v8bf*)(ap1 + c * 32 + 16);
    v16bf a1 = __builtin_shufflevector(h0, h1, 0,1,2,3,4,5,6,7,8,9,10,11,12,13,14,15);
    acc1 = __builtin_amdgcn_wmma_f32_16x16x32_bf16(false, a1, false, b, (short)0, acc1, false, false);
  }

  // D layout: VGPR r -> M = r (lanes 0-15) / 8+r (lanes 16-31); N = lane&15
  const int m0  = (lane < 16) ? 0 : 8;
  const int col = wave * 16 + (lane & 15);
  const float bcol = bias[col];
  #pragma unroll
  for (int r = 0; r < 8; ++r) {
    int row = rowbase0 + m0 + r;
    if (row < Nrows) {
      float v = acc0[r];
      float dv = dinv[row];
      size_t idx = (size_t)row * Nout + col;
      H[idx] = v;
      O[idx] = fmaf(v, dv * dv, bcol);
    }
  }
  #pragma unroll
  for (int r = 0; r < 8; ++r) {
    int row = rowbase1 + m0 + r;
    if (row < Nrows) {
      float v = acc1[r];
      float dv = dinv[row];
      size_t idx = (size_t)row * Nout + col;
      H[idx] = v;
      O[idx] = fmaf(v, dv * dv, bcol);
    }
  }
}

// One wave per edge: out[dst] += h[src] * (dinv[src]*ew*dinv[dst]); F == 128
__global__ void k_edge_scatter(const float* __restrict__ h, float* __restrict__ out,
                               const int* __restrict__ src, const int* __restrict__ dst,
                               const float* __restrict__ ew, const float* __restrict__ dinv,
                               int E, int F) {
  long gt = (long)blockIdx.x * blockDim.x + threadIdx.x;
  int e = (int)(gt >> 5);
  int lane = (int)(gt & 31);
  if (e >= E) return;
  int s = src[e], d = dst[e];
  float nrm = dinv[s] * ew[e] * dinv[d];
  int f = lane * 4;
  if (f >= F) return;
  float4 v = *(const float4*)(h + (size_t)s * F + f);
  float* o = out + (size_t)d * F + f;
  atomicAdd(o + 0, v.x * nrm);
  atomicAdd(o + 1, v.y * nrm);
  atomicAdd(o + 2, v.z * nrm);
  atomicAdd(o + 3, v.w * nrm);
}

// pooled[batch[i]][f] += relu(O[i][f]); thread per (node, 4 features)
__global__ void k_pool(const float* __restrict__ O, const int* __restrict__ batch,
                       float* __restrict__ pooled, long N, int F) {
  long t = (long)blockIdx.x * blockDim.x + threadIdx.x;
  int q4 = F >> 2;
  if (t >= N * q4) return;
  long i = t / q4; int q = (int)(t % q4);
  float4 v = *(const float4*)(O + i * F + q * 4);
  v.x = v.x > 0.f ? v.x : 0.f;  v.y = v.y > 0.f ? v.y : 0.f;
  v.z = v.z > 0.f ? v.z : 0.f;  v.w = v.w > 0.f ? v.w : 0.f;
  float* p = pooled + (size_t)batch[i] * F + q * 4;
  atomicAdd(p + 0, v.x);
  atomicAdd(p + 1, v.y);
  atomicAdd(p + 2, v.z);
  atomicAdd(p + 3, v.w);
}

// Tiny MLP head in fp32: z = relu(pooled@Wm1+bm1); out = z@Wm2+bm2. Block per graph.
__global__ void k_mlp(const float* __restrict__ pooled,
                      const float* __restrict__ Wm1, const float* __restrict__ bm1,
                      const float* __restrict__ Wm2, const float* __restrict__ bm2,
                      float* __restrict__ out, int HID, int OUTD) {
  __shared__ float p[128];
  __shared__ float z[128];
  int g = blockIdx.x, tid = threadIdx.x;
  if (tid < HID) p[tid] = pooled[(size_t)g * HID + tid];
  __syncthreads();
  if (tid < HID) {
    float acc = bm1[tid];
    for (int k = 0; k < HID; ++k) acc = fmaf(p[k], Wm1[(size_t)k * HID + tid], acc);
    z[tid] = acc > 0.f ? acc : 0.f;
  }
  __syncthreads();
  if (tid < OUTD) {
    float acc = bm2[tid];
    for (int k = 0; k < HID; ++k) acc = fmaf(z[k], Wm2[(size_t)k * OUTD + tid], acc);
    out[(size_t)g * OUTD + tid] = acc;
  }
}

extern "C" void kernel_launch(void* const* d_in, const int* in_sizes, int n_in,
                              void* d_out, int out_size, void* d_ws, size_t ws_size,
                              hipStream_t stream) {
  const float* x    = (const float*)d_in[0];
  const int*   ei   = (const int*)d_in[1];
  const int*   batch= (const int*)d_in[2];
  const float* ea   = (const float*)d_in[3];
  const float* W1   = (const float*)d_in[4];
  const float* b1   = (const float*)d_in[5];
  const float* W2   = (const float*)d_in[6];
  const float* b2   = (const float*)d_in[7];
  const float* Wm1  = (const float*)d_in[8];
  const float* bm1  = (const float*)d_in[9];
  const float* Wm2  = (const float*)d_in[10];
  const float* bm2  = (const float*)d_in[11];
  float* out = (float*)d_out;

  const int N    = in_sizes[2];
  const int E    = in_sizes[3];
  const int IN   = in_sizes[0] / N;
  const int HID  = in_sizes[5];
  const int OUTD = in_sizes[11];
  const int G    = out_size / OUTD;

  const int* src = ei;
  const int* dst = ei + E;

  // ---- carve workspace ----
  size_t off = 0;
  auto carve = [&](size_t bytes) -> void* {
    void* p = (char*)d_ws + off;
    off += (bytes + 255) & ~(size_t)255;
    return p;
  };
  float*  dinv   = (float*) carve((size_t)N * sizeof(float));
  __bf16* Ab     = (__bf16*)carve((size_t)N * HID * sizeof(__bf16));   // bf16 activations
  __bf16* Wf     = (__bf16*)carve((size_t)HID * HID * sizeof(__bf16)); // weight fragments
  float*  H      = (float*) carve((size_t)N * HID * sizeof(float));    // GEMM output
  float*  O      = (float*) carve((size_t)N * HID * sizeof(float));    // scatter accumulator
  float*  pooled = (float*) carve((size_t)G * HID * sizeof(float));

  auto cdiv = [](long a, long b) -> int { return (int)((a + b - 1) / b); };

  auto cvt = [&](const float* in, __bf16* outp, long n, int relu) {
    long n8 = n / 8;
    if (n8 > 0)
      k_cvt8_bf16<<<cdiv(n8, BLK), BLK, 0, stream>>>(in, outp, n8, relu);
    long rem = n - n8 * 8;
    if (rem > 0)
      k_cvt_tail_bf16<<<cdiv(rem, BLK), BLK, 0, stream>>>(in, outp, n8 * 8, n, relu);
  };

  const int rowTiles32 = cdiv(N, 32);              // two 16-row M-tiles per block
  const int gemmBlock  = 32 * (HID / 16);          // 256 threads: 8 waves = 8 col tiles

  // 1) degree: 1.0 self-loop + atomic edge-weight accumulation, then rsqrt
  k_fill<<<cdiv(N, BLK), BLK, 0, stream>>>(dinv, 1.0f, N);
  k_deg_accum<<<cdiv(E, BLK), BLK, 0, stream>>>(dinv, dst, ea, E);
  k_dinv<<<cdiv(N, BLK), BLK, 0, stream>>>(dinv, N);

  // 2) layer 1: h = x @ W1 (WMMA bf16), epilogue writes H and O = b1 + h*dinv^2
  cvt(x, Ab, (long)N * IN, 0);
  k_wfrag<<<cdiv(IN * HID, BLK), BLK, 0, stream>>>(W1, Wf, IN, HID);
  k_wmma_gemm_bf16<<<rowTiles32, gemmBlock, 0, stream>>>(Ab, Wf, b1, dinv, H, O, N, IN, HID);
  k_edge_scatter<<<cdiv((long)E * 32, BLK), BLK, 0, stream>>>(H, O, src, dst, ea, dinv, E, HID);

  // 3) layer 2: relu -> bf16, h2 = h1 @ W2, same fused epilogue
  cvt(O, Ab, (long)N * HID, 1);
  k_wfrag<<<cdiv(HID * HID, BLK), BLK, 0, stream>>>(W2, Wf, HID, HID);
  k_wmma_gemm_bf16<<<rowTiles32, gemmBlock, 0, stream>>>(Ab, Wf, b2, dinv, H, O, N, HID, HID);
  k_edge_scatter<<<cdiv((long)E * 32, BLK), BLK, 0, stream>>>(H, O, src, dst, ea, dinv, E, HID);

  // 4) pooling (relu fused) + MLP head
  k_fill<<<cdiv((long)G * HID, BLK), BLK, 0, stream>>>(pooled, 0.0f, (long)G * HID);
  k_pool<<<cdiv((long)N * (HID / 4), BLK), BLK, 0, stream>>>(O, batch, pooled, N, HID);
  k_mlp<<<G, HID, 0, stream>>>(pooled, Wm1, bm1, Wm2, bm2, out, HID, OUTD);

  (void)n_in; (void)ws_size;
}
